// N3Att_6098853560425
// MI455X (gfx1250) — matrix operand order repaired and checked by reading, not compiled, not run on hardware
//
#include <hip/hip_runtime.h>
#include <hip/hip_bf16.h>
#include <cstdint>

// ---------------------------------------------------------------------------
// Types for CDNA5 WMMA (wave32): v16bf = A/B fragment, v8f = C/D fragment
// ---------------------------------------------------------------------------
typedef __attribute__((ext_vector_type(16))) __bf16 v16bf;
typedef __attribute__((ext_vector_type(8)))  float  v8f;

#define DEVINL __device__ __forceinline__

// Async global->LDS copy path (CDNA5 GLOBAL_LOAD_ASYNC_TO_LDS_B128, ASYNCcnt).
#if __has_builtin(__builtin_amdgcn_global_load_async_to_lds_b128) && \
    __has_builtin(__builtin_amdgcn_s_wait_asynccnt)
#define USE_ASYNC_LDS 1
#else
#define USE_ASYNC_LDS 0
#endif

DEVINL uint16_t f2bf(float f) {
    uint32_t u = __float_as_uint(f);
    if ((u & 0x7fffffffu) > 0x7f800000u) return (uint16_t)((u >> 16) | 0x40); // quiet NaN
    uint32_t r = 0x7fffu + ((u >> 16) & 1u);                                  // RNE
    return (uint16_t)((u + r) >> 16);
}

DEVINL float atomic_add_f32(float* p, float v) {
    return __hip_atomic_fetch_add(p, v, __ATOMIC_RELAXED, __HIP_MEMORY_SCOPE_AGENT);
}

#if USE_ASYNC_LDS
// Builtin signature (from compiler diagnostics): 16-byte int-vector pointers,
// arg0 in global (AS1), arg1 in LDS (AS3), imm offset, imm cpol.
typedef int v4i __attribute__((vector_size(16)));
typedef __attribute__((address_space(1))) v4i gv4i;
typedef __attribute__((address_space(3))) v4i sv4i;

// Copy 64 contiguous bytes (32 bf16) global -> LDS via 4 async b128 transfers.
DEVINL void async_cp_64B(const uint16_t* g, uint16_t* l) {
    gv4i* gp = (gv4i*)g;
    sv4i* lp = (sv4i*)l;
    __builtin_amdgcn_global_load_async_to_lds_b128(gp, lp, 0, 0);
    __builtin_amdgcn_global_load_async_to_lds_b128(gp, lp, 16, 0);
    __builtin_amdgcn_global_load_async_to_lds_b128(gp, lp, 32, 0);
    __builtin_amdgcn_global_load_async_to_lds_b128(gp, lp, 48, 0);
}
#define WAIT_ASYNC() __builtin_amdgcn_s_wait_asynccnt(0)
#else
DEVINL void async_cp_64B(const uint16_t* g, uint16_t* l) {
    const uint4* gp = (const uint4*)g;
    *(uint4*)(l + 0)  = gp[0]; *(uint4*)(l + 8)  = gp[1];
    *(uint4*)(l + 16) = gp[2]; *(uint4*)(l + 24) = gp[3];
}
#define WAIT_ASYNC() ((void)0)
#endif

// ---------------------------------------------------------------------------
// bf16 WMMA GEMM:  C[M,N] (f32) = A[M,K] (bf16 row-major) * Bt[N,K] (bf16 row-major)
// (weights pre-transposed so BOTH tiles stage as contiguous K-major copies)
// Block: 256 threads = 8 waves. Block tile 128x128.
// K pipeline: double-buffered LDS (ping-pong), async prefetch of K-step i+1
// overlapped with WMMA compute of K-step i. K must be a multiple of 128.
// Wave tile: 64(M) x 32(N) = 4x2 fragments -> 16 v_wmma per 64-wide K-step.
// ---------------------------------------------------------------------------
__global__ __launch_bounds__(256) void gat_gemm_bf16_wmma(
    const uint16_t* __restrict__ A, const uint16_t* __restrict__ Bt,
    float* __restrict__ C, int M, int N, int K)
{
    constexpr int BUF = 128 * 64;
    __shared__ __align__(16) uint16_t sA[2 * BUF]; // [buf][m][k]
    __shared__ __align__(16) uint16_t sB[2 * BUF]; // [buf][n][k]

    const int tid     = threadIdx.x;
    const int lane    = tid & 31;
    const int wid     = tid >> 5;
    const int lane_lo = lane & 15;
    const int hi      = lane >> 4;           // 0 or 1 (half-wave)
    const int bm      = blockIdx.y * 128;
    const int bn      = blockIdx.x * 128;
    const int wm      = (wid >> 2) * 64;     // wave M offset (0 / 64)
    const int wn      = (wid & 3) * 32;      // wave N offset (0/32/64/96)

    v8f acc[4][2];
#pragma unroll
    for (int mi = 0; mi < 4; ++mi)
#pragma unroll
        for (int ni = 0; ni < 2; ++ni) acc[mi][ni] = (v8f)(0.0f);

    // Staging assignment: thread -> (row 0..127, 32-element K-chunk 0/32)
    const int  srow = tid >> 1;
    const int  koff = (tid & 1) * 32;
    const bool aval = (bm + srow) < M;
    const bool bval = (bn + srow) < N;
    const uint16_t* gA = A  + (size_t)(bm + srow) * K + koff;
    const uint16_t* gB = Bt + (size_t)(bn + srow) * K + koff;
    const int slot = srow * 64 + koff;
    uint16_t* lA0 = &sA[slot];       uint16_t* lA1 = &sA[BUF + slot];
    uint16_t* lB0 = &sB[slot];       uint16_t* lB1 = &sB[BUF + slot];

    // Out-of-range rows never get staged: pre-zero their slots in BOTH buffers.
    {
        const uint4 z = make_uint4(0u, 0u, 0u, 0u);
        if (!aval) {
            *(uint4*)(lA0 + 0) = z; *(uint4*)(lA0 + 8)  = z;
            *(uint4*)(lA0 + 16) = z; *(uint4*)(lA0 + 24) = z;
            *(uint4*)(lA1 + 0) = z; *(uint4*)(lA1 + 8)  = z;
            *(uint4*)(lA1 + 16) = z; *(uint4*)(lA1 + 24) = z;
        }
        if (!bval) {
            *(uint4*)(lB0 + 0) = z; *(uint4*)(lB0 + 8)  = z;
            *(uint4*)(lB0 + 16) = z; *(uint4*)(lB0 + 24) = z;
            *(uint4*)(lB1 + 0) = z; *(uint4*)(lB1 + 8)  = z;
            *(uint4*)(lB1 + 16) = z; *(uint4*)(lB1 + 24) = z;
        }
    }

    // One 64-wide K-step of WMMA compute from buffer base `bb` (0 or BUF).
    auto compute = [&](int bb) {
#pragma unroll
        for (int ks = 0; ks < 2; ++ks) {
            const int kb = ks * 32;
            // Fragment loads per ISA 16-bit A(16x32)/B(32x16) wave32 layouts.
            union F { uint4 q[2]; v16bf v; } af[4], bf[2];
#pragma unroll
            for (int mi = 0; mi < 4; ++mi) {
                int m = wm + mi * 16 + lane_lo;
                af[mi].q[0] = *(const uint4*)&sA[bb + m * 64 + kb + hi * 8];
                af[mi].q[1] = *(const uint4*)&sA[bb + m * 64 + kb + 16 + hi * 8];
            }
#pragma unroll
            for (int ni = 0; ni < 2; ++ni) {
                int n = wn + ni * 16 + lane_lo;
                bf[ni].q[0] = *(const uint4*)&sB[bb + n * 64 + kb + hi * 8];
                bf[ni].q[1] = *(const uint4*)&sB[bb + n * 64 + kb + 16 + hi * 8];
            }
#pragma unroll
            for (int mi = 0; mi < 4; ++mi)
#pragma unroll
                for (int ni = 0; ni < 2; ++ni)
                    acc[mi][ni] = __builtin_amdgcn_wmma_f32_16x16x32_bf16(
                        false, af[mi].v, false, bf[ni].v,
                        (short)0, acc[mi][ni], false, false);
        }
    };

    // Prologue: stage K-step 0 into buffer 0.
    if (aval) async_cp_64B(gA, lA0);
    if (bval) async_cp_64B(gB, lB0);

    for (int k0 = 0; k0 < K; k0 += 128) {
        // Phase 0: consume buf0, prefetch k0+64 into buf1.
        WAIT_ASYNC();
        __syncthreads();     // buf0 landed everywhere; nobody still reads buf1
        if (aval) async_cp_64B(gA + k0 + 64, lA1);
        if (bval) async_cp_64B(gB + k0 + 64, lB1);
        compute(0);

        // Phase 1: consume buf1, prefetch k0+128 into buf0 (if any).
        WAIT_ASYNC();
        __syncthreads();     // buf1 landed everywhere; nobody still reads buf0
        if (k0 + 128 < K) {
            if (aval) async_cp_64B(gA + k0 + 128, lA0);
            if (bval) async_cp_64B(gB + k0 + 128, lB0);
        }
        compute(BUF);
    }

    // Store C (accumulator layout: VGPR j -> row j + 8*hi, col = lane%16)
#pragma unroll
    for (int mi = 0; mi < 4; ++mi) {
#pragma unroll
        for (int ni = 0; ni < 2; ++ni) {
            int c = bn + wn + ni * 16 + lane_lo;
#pragma unroll
            for (int j = 0; j < 8; ++j) {
                int r = bm + wm + mi * 16 + hi * 8 + j;
                if (r < M && c < N) C[(size_t)r * N + c] = acc[mi][ni][j];
            }
        }
    }
}

// ---------------------------------------------------------------------------
// Attention helpers
// ---------------------------------------------------------------------------
__global__ __launch_bounds__(256) void gat_node_scores(
    const float* __restrict__ h, const float* __restrict__ a_s,
    const float* __restrict__ a_d, float* __restrict__ s_s,
    float* __restrict__ s_d, int F, int stride)
{
    int node = blockIdx.x, tid = threadIdx.x;
    const float* row = h + (size_t)node * stride;
    float p1 = 0.f, p2 = 0.f;
    for (int f = tid; f < F; f += 256) {
        float v = row[f];
        p1 += v * a_s[f];
        p2 += v * a_d[f];
    }
    __shared__ float red[512];
    red[tid] = p1; red[256 + tid] = p2;
    __syncthreads();
    for (int s = 128; s > 0; s >>= 1) {
        if (tid < s) { red[tid] += red[tid + s]; red[256 + tid] += red[256 + tid + s]; }
        __syncthreads();
    }
    if (tid == 0) { s_s[node] = red[0]; s_d[node] = red[256]; }
}

__global__ void gat_edge_logits(const int* __restrict__ ei, const float* __restrict__ s_s,
                                const float* __restrict__ s_d, float* __restrict__ e_buf,
                                unsigned int* __restrict__ m_u, int nE, int nN)
{
    int e = blockIdx.x * blockDim.x + threadIdx.x;
    if (e >= nE + nN) return;
    int s, d;
    if (e < nE) { s = ei[e]; d = ei[nE + e]; } else { s = e - nE; d = s; }
    float v = s_s[s] + s_d[d];
    v = v > 0.f ? v : 0.2f * v;                       // leaky_relu(0.2)
    e_buf[e] = v;
    uint32_t u = __float_as_uint(v);
    uint32_t k = (u & 0x80000000u) ? ~u : (u | 0x80000000u); // order-preserving map
    atomicMax(&m_u[d], k);
}

__global__ void gat_edge_exp(const int* __restrict__ ei, const float* __restrict__ e_buf,
                             const unsigned int* __restrict__ m_u, float* __restrict__ ex_buf,
                             float* __restrict__ denom, int nE, int nN)
{
    int e = blockIdx.x * blockDim.x + threadIdx.x;
    if (e >= nE + nN) return;
    int d = (e < nE) ? ei[nE + e] : (e - nE);
    uint32_t k = m_u[d];
    uint32_t u = (k & 0x80000000u) ? (k & 0x7fffffffu) : ~k;
    float m = __uint_as_float(u);
    float ex = __expf(e_buf[e] - m);
    ex_buf[e] = ex;
    atomic_add_f32(&denom[d], ex);
}

__global__ void gat_edge_agg(const int* __restrict__ ei, const float* __restrict__ h,
                             const float* __restrict__ ex_buf, const float* __restrict__ denom,
                             float* __restrict__ out, int nE, int nN, int F, int stride)
{
    int e = blockIdx.x;
    int s, d;
    if (e < nE) { s = ei[e]; d = ei[nE + e]; } else { s = e - nE; d = s; }
    float alpha = ex_buf[e] / denom[d];
    const float* hs = h + (size_t)s * stride;
    float* od = out + (size_t)d * stride;
    for (int f = threadIdx.x; f < F; f += blockDim.x)
        atomic_add_f32(&od[f], alpha * hs[f]);
}

__global__ void gat_bias_relu_bf16(const float* __restrict__ in, const float* __restrict__ b,
                                   uint16_t* __restrict__ out, int n, int F)
{
    int i = blockIdx.x * blockDim.x + threadIdx.x;
    if (i >= n) return;
    float v = in[i] + b[i % F];
    v = v > 0.f ? v : 0.f;
    out[i] = f2bf(v);
}

__global__ void gat_logsoftmax7(const float* __restrict__ in, const float* __restrict__ b,
                                float* __restrict__ out, int n)
{
    int i = blockIdx.x * blockDim.x + threadIdx.x;
    if (i >= n) return;
    float v[7];
    float m = -1e30f;
#pragma unroll
    for (int f = 0; f < 7; ++f) { v[f] = in[i * 16 + f] + b[f]; m = fmaxf(m, v[f]); }
    float s = 0.f;
#pragma unroll
    for (int f = 0; f < 7; ++f) s += __expf(v[f] - m);
    float ls = m + __logf(s);
#pragma unroll
    for (int f = 0; f < 7; ++f) out[i * 7 + f] = v[f] - ls;
}

__global__ void cvt_f32_to_bf16(const float* __restrict__ src, uint16_t* __restrict__ dst,
                                size_t n)
{
    size_t i = (size_t)blockIdx.x * blockDim.x + threadIdx.x;
    if (i < n) dst[i] = f2bf(src[i]);
}

// W (K x N f32, row-major) -> Wt (N x K bf16, row-major)
__global__ void cvt_w_t_bf16(const float* __restrict__ src, uint16_t* __restrict__ dst,
                             int K, int N)
{
    int i = blockIdx.x * blockDim.x + threadIdx.x;
    if (i >= K * N) return;
    int n = i / K, k = i - n * K;          // dst index i = n*K + k
    dst[i] = f2bf(src[(size_t)k * N + n]);
}

// W3 (K x 7 f32) -> W3t (16 x K bf16), rows 7..15 zero
__global__ void cvt_w3_t_pad(const float* __restrict__ src, uint16_t* __restrict__ dst, int K)
{
    int i = blockIdx.x * blockDim.x + threadIdx.x;
    if (i >= 16 * K) return;
    int n = i / K, k = i - n * K;
    dst[i] = (n < 7) ? f2bf(src[(size_t)k * 7 + n]) : (uint16_t)0;
}

// ---------------------------------------------------------------------------
// Host orchestration
// ---------------------------------------------------------------------------
extern "C" void kernel_launch(void* const* d_in, const int* in_sizes, int n_in,
                              void* d_out, int out_size, void* d_ws, size_t ws_size,
                              hipStream_t stream)
{
    (void)in_sizes; (void)n_in; (void)out_size; (void)ws_size;
    const int M = 10000, E = 80000;
    const int IN = 512, H1D = 4096, H2D = 1024, OUTP = 16;
    const int E2 = E + M;

    const float* x   = (const float*)d_in[0];
    const int*   ei  = (const int*)  d_in[1];
    const float* W1  = (const float*)d_in[2];
    const float* as1 = (const float*)d_in[3];
    const float* ad1 = (const float*)d_in[4];
    const float* b1  = (const float*)d_in[5];
    const float* W2  = (const float*)d_in[6];
    const float* as2 = (const float*)d_in[7];
    const float* ad2 = (const float*)d_in[8];
    const float* b2  = (const float*)d_in[9];
    const float* W3  = (const float*)d_in[10];
    const float* as3 = (const float*)d_in[11];
    const float* ad3 = (const float*)d_in[12];
    const float* b3  = (const float*)d_in[13];

    char* ws = (char*)d_ws;
    size_t off = 0;
    auto walloc = [&](size_t bytes) -> void* {
        void* p = ws + off;
        off = (off + bytes + 255) & ~(size_t)255;
        return p;
    };
    float*    Abuf = (float*)   walloc((size_t)M * H1D * 4);  // GEMM output h
    uint16_t* Bbuf = (uint16_t*)walloc((size_t)M * H1D * 2);  // bf16 layer input
    float*    Cbuf = (float*)   walloc((size_t)M * H1D * 4);  // aggregated output
    uint16_t* W1t  = (uint16_t*)walloc((size_t)H1D * IN  * 2);
    uint16_t* W2t  = (uint16_t*)walloc((size_t)H2D * H1D * 2);
    uint16_t* W3t  = (uint16_t*)walloc((size_t)OUTP * H2D * 2);
    float*        s_s   = (float*)       walloc((size_t)M * 4);
    float*        s_d   = (float*)       walloc((size_t)M * 4);
    unsigned int* m_u   = (unsigned int*)walloc((size_t)M * 4);
    float*        denom = (float*)       walloc((size_t)M * 4);
    float*        e_buf = (float*)       walloc((size_t)E2 * 4);
    float*        exb   = (float*)       walloc((size_t)E2 * 4);

    // ---- convert inputs to bf16 (weights transposed to [N][K]) ----
    {
        size_t n = (size_t)M * IN;
        cvt_f32_to_bf16<<<(unsigned)((n + 255) / 256), 256, 0, stream>>>(x, Bbuf, n);
    }
    cvt_w_t_bf16<<<(IN * H1D + 255) / 256, 256, 0, stream>>>(W1, W1t, IN, H1D);
    cvt_w_t_bf16<<<(H1D * H2D + 255) / 256, 256, 0, stream>>>(W2, W2t, H1D, H2D);
    cvt_w3_t_pad<<<(16 * H2D + 255) / 256, 256, 0, stream>>>(W3, W3t, H2D);

    auto layer = [&](int K, const uint16_t* Wt, int Np, int F,
                     const float* a_s, const float* a_d, int aggBlock) {
        dim3 grid((Np + 127) / 128, (M + 127) / 128);
        gat_gemm_bf16_wmma<<<grid, 256, 0, stream>>>(Bbuf, Wt, Abuf, M, Np, K);
        gat_node_scores<<<M, 256, 0, stream>>>(Abuf, a_s, a_d, s_s, s_d, F, Np);
        (void)hipMemsetAsync(m_u, 0, (size_t)M * 4, stream);
        (void)hipMemsetAsync(denom, 0, (size_t)M * 4, stream);
        (void)hipMemsetAsync(Cbuf, 0, (size_t)M * Np * 4, stream);
        gat_edge_logits<<<(E2 + 255) / 256, 256, 0, stream>>>(ei, s_s, s_d, e_buf, m_u, E, M);
        gat_edge_exp<<<(E2 + 255) / 256, 256, 0, stream>>>(ei, e_buf, m_u, exb, denom, E, M);
        gat_edge_agg<<<E2, aggBlock, 0, stream>>>(ei, Abuf, exb, denom, Cbuf, E, M, F, Np);
    };

    // Layer 1: 512 -> 4096, ReLU
    layer(IN, W1t, H1D, H1D, as1, ad1, 256);
    {
        int n = M * H1D;
        gat_bias_relu_bf16<<<(n + 255) / 256, 256, 0, stream>>>(Cbuf, b1, Bbuf, n, H1D);
    }
    // Layer 2: 4096 -> 1024, ReLU
    layer(H1D, W2t, H2D, H2D, as2, ad2, 256);
    {
        int n = M * H2D;
        gat_bias_relu_bf16<<<(n + 255) / 256, 256, 0, stream>>>(Cbuf, b2, Bbuf, n, H2D);
    }
    // Layer 3: 1024 -> 7 (padded to 16), log_softmax
    layer(H2D, W3t, OUTP, 7, as3, ad3, 32);
    gat_logsoftmax7<<<(M + 255) / 256, 256, 0, stream>>>(Cbuf, b3, (float*)d_out, M);
}
